// NonlocalBlock_4131758539440
// MI455X (gfx1250) — compile-verified, matrix operand order checked by reading, not run on hardware
//
#include <hip/hip_runtime.h>

typedef __attribute__((ext_vector_type(16))) __bf16 v16bf;
typedef __attribute__((ext_vector_type(8)))  float  v8f;
typedef int v4i32 __attribute__((vector_size(16)));

#define IN_CH     256
#define VC        128
#define KC        64
#define MSG       2
#define BATCH     4
#define NPIX      4096          // 64*64
#define PROJ_ROWS 512           // 256 (wv) + 128 (wk) + 128 (rk)

#define AS3 __attribute__((address_space(3)))

#if defined(__has_builtin)
#if __has_builtin(__builtin_amdgcn_global_load_async_to_lds_b128) && \
    __has_builtin(__builtin_amdgcn_s_wait_asynccnt)
#define HAVE_ASYNC 1
#endif
#endif
#ifndef HAVE_ASYNC
#define HAVE_ASYNC 0
#endif

// K index inside a 16x32 bf16 A/B tile for packed element e (0..15) of a lane.
// (ISA 7.12.2): VGPR group 0-3 holds K=0..7 (lanes 0-15) / K=8..15 (lanes 16-31),
// VGPR group 4-7 holds K=16..23 / K=24..31.  => k(e) = (e>>3)*16 + lhi*8 + (e&7)
__device__ __forceinline__ int kmap(int e, int lhi) {
    return ((e >> 3) << 4) + (lhi << 3) + (e & 7);
}

__device__ __forceinline__ v8f wmma_bf16(v16bf a, v16bf b, v8f c) {
    return __builtin_amdgcn_wmma_f32_16x16x32_bf16(
        /*neg_a=*/false, a, /*neg_b=*/false, b,
        /*c_mod=*/(short)0, c, /*reuse_a=*/false, /*reuse_b=*/false);
}

// Build a v16bf operand from two contiguous 16-byte chunks (elements 0-7 from
// base0, 8-15 from base1) -> two ds_load_b128 / global_load_b128.
__device__ __forceinline__ v16bf ld_tile(const __bf16* base0, const __bf16* base1) {
    v16bf r;
    *(uint4*)&r       = *(const uint4*)base0;
    *((uint4*)&r + 1) = *(const uint4*)base1;
    return r;
}

// ---------------------------------------------------------------------------
// Kernel 1: fused projections. proj[b][o][n] (bf16), o: 0-255=wv, 256-383=wk,
// 384-511=rk. One wave computes one 16x16 tile over K=IN_CH.
// ---------------------------------------------------------------------------
__global__ void proj_kernel(const float* __restrict__ x,
                            const float* __restrict__ wv_w, const float* __restrict__ wv_b,
                            const float* __restrict__ wk_w, const float* __restrict__ wk_b,
                            const float* __restrict__ rk_w, const float* __restrict__ rk_b,
                            __bf16* __restrict__ proj) {
    const int lane = threadIdx.x & 31;
    const int wave = threadIdx.x >> 5;
    const int l15  = lane & 15;
    const int lhi  = lane >> 4;
    const int gid  = blockIdx.x * 8 + wave;        // 4*32*256 = 32768 tiles
    const int nt = gid & 255;
    const int ot = (gid >> 8) & 31;
    const int b  = gid >> 13;
    const int o0 = ot * 16;
    const int n0 = nt * 16;

    const float* W; const float* Bv; int rowoff;
    if (o0 < 256)      { W = wv_w; Bv = wv_b; rowoff = o0; }
    else if (o0 < 384) { W = wk_w; Bv = wk_b; rowoff = o0 - 256; }
    else               { W = rk_w; Bv = rk_b; rowoff = o0 - 384; }

    const float* Arow = W + (size_t)(rowoff + l15) * IN_CH + 8 * lhi;
    const float* xb   = x + (size_t)b * IN_CH * NPIX + n0 + l15;

    v8f acc = {};
    for (int k0 = 0; k0 < IN_CH; k0 += 32) {
        // A: 4x global_load_b128 along the contiguous K rows of W.
        float4 f0 = *(const float4*)(Arow + k0);
        float4 f1 = *(const float4*)(Arow + k0 + 4);
        float4 f2 = *(const float4*)(Arow + k0 + 16);
        float4 f3 = *(const float4*)(Arow + k0 + 20);
        float fa[16] = {f0.x, f0.y, f0.z, f0.w, f1.x, f1.y, f1.z, f1.w,
                        f2.x, f2.y, f2.z, f2.w, f3.x, f3.y, f3.z, f3.w};
        v16bf a, bb;
        #pragma unroll
        for (int e = 0; e < 16; ++e) {
            a[e]  = (__bf16)fa[e];
            bb[e] = (__bf16)xb[(size_t)(k0 + kmap(e, lhi)) * NPIX];
        }
        acc = wmma_bf16(a, bb, acc);
    }
    __bf16* dst = proj + (size_t)b * PROJ_ROWS * NPIX;
    #pragma unroll
    for (int r = 0; r < 8; ++r) {
        int ol = r + 8 * lhi;
        float v = acc[r] + Bv[rowoff + ol];
        dst[(size_t)(o0 + ol) * NPIX + n0 + l15] = (__bf16)v;
    }
}

// ---------------------------------------------------------------------------
// Kernel 2: streaming attention. Block = 8 waves = 128 q columns of one
// batch-message; loop p in steps of 32, accumulating numerator (WMMA) and
// denominator (softplus row sums). Never materializes the 4096x4096 matrix.
// ---------------------------------------------------------------------------
__global__ void attn_kernel(const __bf16* __restrict__ proj,
                            __bf16* __restrict__ readBuf) {
    const int tid  = threadIdx.x;
    const int lane = tid & 31;
    const int wave = tid >> 5;
    const int l15  = lane & 15;
    const int lhi  = lane >> 4;
    const int bm   = blockIdx.y;                   // 0..7
    const int b    = bm >> 1;
    const int mm   = bm & 1;
    const int q0   = blockIdx.x * 128 + wave * 16; // 32 blocks in x

    __shared__ __align__(16) __bf16 s_wv[128 * 32];  // row-major [v][p]   8 KB
    __shared__ __align__(16) __bf16 s_wkT[32 * 64];  // transposed [p][k]  4 KB
    __shared__ __align__(16) __bf16 s_S[8][16 * 32]; // per-wave S [q][p]  8 KB
    __shared__ float s_den[8][16];

    const __bf16* pb     = proj + (size_t)b * PROJ_ROWS * NPIX;
    const __bf16* wkRows = pb + (size_t)(256 + mm * KC) * NPIX;
    const __bf16* wvRows = pb + (size_t)(mm * VC) * NPIX;
    const __bf16* rkRows = pb + (size_t)(384 + mm * KC) * NPIX;

    // Preload A = rk^T for this q-tile: two 16x32 K-halves (K=64 total).
    v16bf aS0, aS1;
    #pragma unroll
    for (int e = 0; e < 16; ++e) {
        int k = kmap(e, lhi);
        aS0[e] = rkRows[(size_t)k        * NPIX + q0 + l15];
        aS1[e] = rkRows[(size_t)(32 + k) * NPIX + q0 + l15];
    }

    v8f num[8];
    #pragma unroll
    for (int t = 0; t < 8; ++t) num[t] = (v8f){};
    float dsum[8];
    #pragma unroll
    for (int r = 0; r < 8; ++r) dsum[r] = 0.f;

    __bf16* myS = s_S[wave];

    for (int p0 = 0; p0 < NPIX; p0 += 32) {
        __syncthreads();
        // --- Stage wv 128x32 row-major: 512 x 16-byte segments ---------------
        #pragma unroll
        for (int i = 0; i < 2; ++i) {
            int idx = tid + i * 256;               // 0..511
            int row = idx >> 2, seg = idx & 3;
            const __bf16* g = wvRows + (size_t)row * NPIX + p0 + seg * 8;
            __bf16* l = s_wv + row * 32 + seg * 8;
#if HAVE_ASYNC
            __builtin_amdgcn_global_load_async_to_lds_b128(
                (v4i32*)g, (AS3 v4i32*)l, 0, 0);
#else
            *(uint4*)l = *(const uint4*)g;
#endif
        }
        // --- Stage wk 64x32 transposed to [p][k] (async can't transpose) -----
        #pragma unroll
        for (int i = 0; i < 4; ++i) {
            int idx = tid + i * 256;               // 0..1023
            int kk = idx >> 4, cu = idx & 15;
            unsigned int u = *(const unsigned int*)(wkRows + (size_t)kk * NPIX + p0 + cu * 2);
            const __bf16* h = (const __bf16*)&u;
            s_wkT[(2 * cu)     * 64 + kk] = h[0];
            s_wkT[(2 * cu + 1) * 64 + kk] = h[1];
        }
        // Prefetch next p-tile while this one is consumed.
        if (p0 + 32 < NPIX) {
            if (tid < 128) __builtin_prefetch(wvRows + (size_t)tid * NPIX + p0 + 32, 0, 3);
            else __builtin_prefetch(wkRows + (size_t)((tid - 128) & 63) * NPIX + p0 + 32, 0, 3);
        }
#if HAVE_ASYNC
        __builtin_amdgcn_s_wait_asynccnt(0);
#endif
        __syncthreads();

        // --- S[q, p0+0..31] = softplus(rk^T wk): 2 p-subtiles, K=64 ----------
        #pragma unroll
        for (int j2 = 0; j2 < 2; ++j2) {
            const __bf16* col = s_wkT + (j2 * 16 + l15) * 64 + 8 * lhi;
            v16bf b0 = ld_tile(col,      col + 16);   // kk = 0..31 half
            v16bf b1 = ld_tile(col + 32, col + 48);   // kk = 32..63 half
            v8f s = (v8f){};
            s = wmma_bf16(aS0, b0, s);
            s = wmma_bf16(aS1, b1, s);
            #pragma unroll
            for (int r = 0; r < 8; ++r) {
                float v = s[r];                       // S[q=r+8*lhi][p=16*j2+l15]
                v = (v > 20.f) ? v : log1pf(__expf(v));
                dsum[r] += v;
                myS[(r + 8 * lhi) * 32 + j2 * 16 + l15] = (__bf16)v;
            }
        }
        // --- num[v,q] += wv[v, p-tile] * S^T (same-wave LDS ops in-order) ----
        const __bf16* srow = myS + l15 * 32 + 8 * lhi;
        v16bf bs = ld_tile(srow, srow + 16);
        #pragma unroll
        for (int t = 0; t < 8; ++t) {
            const __bf16* arow = s_wv + (t * 16 + l15) * 32 + 8 * lhi;
            v16bf a = ld_tile(arow, arow + 16);
            num[t] = wmma_bf16(a, bs, num[t]);
        }
    }

    // Denominator: reduce p-partials across the 16 lanes sharing a q row.
    #pragma unroll
    for (int r = 0; r < 8; ++r) {
        float v = dsum[r];
        v += __shfl_xor(v, 1, 32);
        v += __shfl_xor(v, 2, 32);
        v += __shfl_xor(v, 4, 32);
        v += __shfl_xor(v, 8, 32);
        if (l15 == 0) s_den[wave][8 * lhi + r] = v + 1e-4f;
    }
    float dq = s_den[wave][l15];   // same-wave LDS in-order: store precedes load

    __bf16* rb = readBuf + (size_t)b * (VC * MSG) * NPIX;
    #pragma unroll
    for (int t = 0; t < 8; ++t) {
        #pragma unroll
        for (int r = 0; r < 8; ++r) {
            int v = t * 16 + r + 8 * lhi;            // v_local 0..127
            int c = mm * VC + v;
            rb[(size_t)c * NPIX + q0 + l15] = (__bf16)(num[t][r] / dq);
        }
    }
}

// ---------------------------------------------------------------------------
// Kernel 3: out = x*shortcut + (rv_w @ read + rv_b)*residue. One wave per
// 16x16 tile, K=256 over bf16 read buffer, fp32 output.
// ---------------------------------------------------------------------------
__global__ void out_kernel(const float* __restrict__ x,
                           const float* __restrict__ rv_w, const float* __restrict__ rv_b,
                           const float* __restrict__ shortcut, const float* __restrict__ residue,
                           const __bf16* __restrict__ readBuf, float* __restrict__ out) {
    const int lane = threadIdx.x & 31;
    const int wave = threadIdx.x >> 5;
    const int l15  = lane & 15;
    const int lhi  = lane >> 4;
    const int gid  = blockIdx.x * 8 + wave;        // 4*16*256 = 16384 tiles
    const int nt = gid & 255;
    const int ot = (gid >> 8) & 15;
    const int b  = gid >> 12;
    const int o0 = ot * 16;
    const int n0 = nt * 16;

    const float*  Arow = rv_w + (size_t)(o0 + l15) * IN_CH + 8 * lhi;
    const __bf16* rbB  = readBuf + (size_t)b * IN_CH * NPIX + n0 + l15;

    v8f acc = {};
    for (int k0 = 0; k0 < IN_CH; k0 += 32) {
        float4 f0 = *(const float4*)(Arow + k0);
        float4 f1 = *(const float4*)(Arow + k0 + 4);
        float4 f2 = *(const float4*)(Arow + k0 + 16);
        float4 f3 = *(const float4*)(Arow + k0 + 20);
        float fa[16] = {f0.x, f0.y, f0.z, f0.w, f1.x, f1.y, f1.z, f1.w,
                        f2.x, f2.y, f2.z, f2.w, f3.x, f3.y, f3.z, f3.w};
        v16bf a, bb;
        #pragma unroll
        for (int e = 0; e < 16; ++e) {
            a[e]  = (__bf16)fa[e];
            bb[e] = rbB[(size_t)(k0 + kmap(e, lhi)) * NPIX];
        }
        acc = wmma_bf16(a, bb, acc);
    }
    #pragma unroll
    for (int r = 0; r < 8; ++r) {
        int o = o0 + r + 8 * lhi;
        size_t xi = ((size_t)b * IN_CH + o) * NPIX + n0 + l15;
        out[xi] = x[xi] * shortcut[o] + (acc[r] + rv_b[o]) * residue[o];
    }
}

// ---------------------------------------------------------------------------
extern "C" void kernel_launch(void* const* d_in, const int* in_sizes, int n_in,
                              void* d_out, int out_size, void* d_ws, size_t ws_size,
                              hipStream_t stream) {
    const float* x        = (const float*)d_in[0];
    const float* wv_w     = (const float*)d_in[1];
    const float* wv_b     = (const float*)d_in[2];
    const float* wk_w     = (const float*)d_in[3];
    const float* wk_b     = (const float*)d_in[4];
    const float* rk_w     = (const float*)d_in[5];
    const float* rk_b     = (const float*)d_in[6];
    const float* rv_w     = (const float*)d_in[7];
    const float* rv_b     = (const float*)d_in[8];
    const float* shortcut = (const float*)d_in[9];
    const float* residue  = (const float*)d_in[10];

    __bf16* proj    = (__bf16*)d_ws;                              // 4*512*4096 bf16 = 16 MB
    __bf16* readBuf = proj + (size_t)BATCH * PROJ_ROWS * NPIX;    // 4*256*4096 bf16 =  8 MB

    proj_kernel<<<4096, 256, 0, stream>>>(x, wv_w, wv_b, wk_w, wk_b, rk_w, rk_b, proj);
    attn_kernel<<<dim3(32, 8), 256, 0, stream>>>(proj, readBuf);
    out_kernel<<<2048, 256, 0, stream>>>(x, rv_w, rv_b, shortcut, residue, readBuf,
                                         (float*)d_out);
}
